// MultiHeadAttention_78864189489198
// MI455X (gfx1250) — matrix operand order repaired and checked by reading, not compiled, not run on hardware
//
#include <hip/hip_runtime.h>
#include <hip/hip_bf16.h>

// Problem constants (match reference): B=2, N=2048, D=2048, H=16, HD=128
#define BDIM   2
#define NSEQ   2048
#define DMODEL 2048
#define NHEADS 16
#define HDIM   128

typedef __bf16 bf16_t;
typedef __attribute__((ext_vector_type(16))) __bf16 v16bf;
typedef __attribute__((ext_vector_type(8)))  __bf16 v8bf;
typedef __attribute__((ext_vector_type(4)))  __bf16 v4bf;
typedef __attribute__((ext_vector_type(8)))  float  v8f;

// D = A(16x32 bf16) * B(32x16 bf16) + C(16x16 f32), wave32.
static __device__ __forceinline__ v8f wmma_bf16(v16bf a, v16bf b, v8f c) {
  return __builtin_amdgcn_wmma_f32_16x16x32_bf16(
      /*neg_a=*/false, a, /*neg_b=*/false, b,
      /*c_mod=*/(short)0, c, /*reuse_a=*/false, /*reuse_b=*/false);
}

// Join two 16-byte (8 x bf16) chunks into one v16bf fragment register set.
static __device__ __forceinline__ v16bf join8(const bf16_t* p0, const bf16_t* p1) {
  v8bf lo = *(const v8bf*)p0;
  v8bf hi = *(const v8bf*)p1;
  return __builtin_shufflevector(lo, hi, 0,1,2,3,4,5,6,7,8,9,10,11,12,13,14,15);
}

// 4-element loaders (overloaded on source type), converting to bf16.
static __device__ __forceinline__ v4bf cvt4(const float* p) {
  float4 v = *(const float4*)p;
  v4bf d; d[0] = (bf16_t)v.x; d[1] = (bf16_t)v.y;
  d[2] = (bf16_t)v.z; d[3] = (bf16_t)v.w;
  return d;
}
static __device__ __forceinline__ v4bf cvt4(const bf16_t* p) {
  return *(const v4bf*)p;
}

// ---------------------------------------------------------------------------
// GEMM: C[M,Nd] = A[M,K] * W[K,Nd] (+ bias).  A is fp32 or bf16; W fp32.
// 128x128 tile per block, 256 threads (8 waves, 4x2 wave grid), double-
// buffered 32-wide K slices, 8 v_wmma_f32_16x16x32_bf16 per slice per wave.
// ---------------------------------------------------------------------------
template<typename AT, typename OT, bool BIAS>
__global__ __launch_bounds__(256)
void gemm_wmma(const AT* __restrict__ A, const float* __restrict__ W,
               const float* __restrict__ bias, OT* __restrict__ C,
               int M, int Nd, int K)
{
  __shared__ bf16_t As[2][128][40];   // [m][k]   ping-pong
  __shared__ bf16_t Bs[2][128][40];   // [n][k]   (W tile stored transposed)

  const int tid  = threadIdx.x;
  const int lane = tid & 31;
  const int wave = tid >> 5;
  const int hi   = lane >> 4;      // half-wave select
  const int ln   = lane & 15;

  const int bm = blockIdx.y * 128;
  const int bn = blockIdx.x * 128;
  const int wm = (wave & 3) * 32;  // wave row offset within tile
  const int wn = (wave >> 2) * 64; // wave col offset within tile

  // Staging thread mapping
  const int ar0 = tid >> 3;        // A tile: row base 0..31 (x4 passes)
  const int ac4 = (tid & 7) * 4;   // A tile: k column 0..28
  const int wr0 = tid >> 5;        // W tile: k row base 0..7 (x4 passes)
  const int wc4 = (tid & 31) * 4;  // W tile: n column 0..124

  v8f acc[2][4] = {};

  v4bf   aReg[4];
  float4 wReg[4];

  auto globalLoad = [&](int k0) {
    #pragma unroll
    for (int rr = 0; rr < 4; ++rr)
      aReg[rr] = cvt4(A + (size_t)(bm + ar0 + rr * 32) * K + k0 + ac4);
    #pragma unroll
    for (int rr = 0; rr < 4; ++rr)
      wReg[rr] = *(const float4*)(W + (size_t)(k0 + wr0 + rr * 8) * Nd + bn + wc4);
  };
  auto ldsStore = [&](int buf) {
    #pragma unroll
    for (int rr = 0; rr < 4; ++rr)
      *(v4bf*)&As[buf][ar0 + rr * 32][ac4] = aReg[rr];
    #pragma unroll
    for (int rr = 0; rr < 4; ++rr) {
      const int krow = wr0 + rr * 8;
      Bs[buf][wc4 + 0][krow] = (bf16_t)wReg[rr].x;
      Bs[buf][wc4 + 1][krow] = (bf16_t)wReg[rr].y;
      Bs[buf][wc4 + 2][krow] = (bf16_t)wReg[rr].z;
      Bs[buf][wc4 + 3][krow] = (bf16_t)wReg[rr].w;
    }
  };

  globalLoad(0);
  ldsStore(0);
  __syncthreads();

  const int T = K / 32;
  for (int t = 0; t < T; ++t) {
    const int cur = t & 1;

    if (t + 2 < T) {   // prefetch two slices ahead -> global_prefetch_b8
      __builtin_prefetch(A + (size_t)(bm + ar0) * K + (t + 2) * 32 + ac4, 0, 1);
      __builtin_prefetch(W + (size_t)((t + 2) * 32 + wr0) * Nd + bn + wc4, 0, 1);
    }
    if (t + 1 < T) globalLoad((t + 1) * 32);

    // Fragments per ISA layouts:
    //  A (16x32): lane hi=0 -> K {0..7}u{16..23}; hi=1 -> K {8..15}u{24..31}
    //  B (32x16): same K split as A: hi=0 -> K {0..7}u{16..23}; hi=1 -> K {8..15}u{24..31}
    v16bf af[2], bf4[4];
    #pragma unroll
    for (int i = 0; i < 2; ++i) {
      const bf16_t* ap = &As[cur][wm + i * 16 + ln][0];
      af[i] = join8(ap + hi * 8, ap + 16 + hi * 8);
    }
    #pragma unroll
    for (int j = 0; j < 4; ++j) {
      const bf16_t* bp = &Bs[cur][wn + j * 16 + ln][0];
      bf4[j] = join8(bp + hi * 8, bp + 16 + hi * 8);
    }
    #pragma unroll
    for (int i = 0; i < 2; ++i)
      #pragma unroll
      for (int j = 0; j < 4; ++j)
        acc[i][j] = wmma_bf16(af[i], bf4[j], acc[i][j]);

    if (t + 1 < T) ldsStore(cur ^ 1);   // other buffer: safe vs readers of cur
    __syncthreads();
  }

  // Epilogue: C layout — VGPR r: row = r + 8*hi, col = ln (per 16x16 frag)
  #pragma unroll
  for (int i = 0; i < 2; ++i)
    #pragma unroll
    for (int j = 0; j < 4; ++j)
      #pragma unroll
      for (int r = 0; r < 8; ++r) {
        const int grow = bm + wm + i * 16 + r + hi * 8;
        const int gcol = bn + wn + j * 16 + ln;
        float v = acc[i][j][r];
        if (BIAS) v += bias[gcol];
        C[(size_t)grow * Nd + gcol] = (OT)v;
      }
}

// ---------------------------------------------------------------------------
// Flash-style attention: per (b,h), per 128-row query block (8 waves x 16
// rows). Double-buffered 32-key tiles through LDS; online softmax; never
// materializes the NxN score matrix.
// ---------------------------------------------------------------------------
__global__ __launch_bounds__(256)
void attn_wmma(const bf16_t* __restrict__ Q, bf16_t* __restrict__ Y)
{
  __shared__ bf16_t Kt [2][32][136];   // key tile   [key][hd]  (S: B = Q^T)
  __shared__ bf16_t Ktt[2][128][40];   // transposed [hd][key]  (Y: B = Q)
  __shared__ bf16_t Pl [8][16][32];    // per-wave P tile (C->A relayout)

  const int tid  = threadIdx.x;
  const int lane = tid & 31;
  const int wave = tid >> 5;
  const int hi   = lane >> 4;
  const int ln   = lane & 15;

  const int bh = blockIdx.x;
  const int b  = bh / NHEADS;
  const int h  = bh % NHEADS;
  const size_t base = (size_t)b * NSEQ * DMODEL + (size_t)h * HDIM;
  const bf16_t* qb = Q + base;
  bf16_t*       yb = Y + base;

  const int m0 = blockIdx.y * 128 + wave * 16;   // this wave's query rows

  // Q A-fragments for the wave's 16 rows, all 4 K-steps (hd 0..127): load once
  v16bf aq[4];
  {
    const bf16_t* qrow = qb + (size_t)(m0 + ln) * DMODEL;
    #pragma unroll
    for (int ks = 0; ks < 4; ++ks)
      aq[ks] = join8(qrow + ks * 32 + hi * 8, qrow + ks * 32 + 16 + hi * 8);
  }

  // Key-tile staging (double buffered)
  const int krow = tid & 31;                  // 0..31
  const int cb   = (tid >> 5) * 16;           // 0..112
  v8bf sa, sc;
  auto kload = [&](int kb) {
    const bf16_t* src = qb + (size_t)(kb + krow) * DMODEL + cb;
    sa = *(const v8bf*)src;
    sc = *(const v8bf*)(src + 8);
  };
  auto kstore = [&](int buf) {
    *(v8bf*)&Kt[buf][krow][cb]     = sa;
    *(v8bf*)&Kt[buf][krow][cb + 8] = sc;
    #pragma unroll
    for (int j = 0; j < 8; ++j) {
      Ktt[buf][cb + j][krow]     = sa[j];
      Ktt[buf][cb + 8 + j][krow] = sc[j];
    }
  };

  const float scale = 0.08838834764831845f;      // 1/sqrt(128)
  float rowmax[8], rowsum[8];
  v8f   acc[8] = {};
  #pragma unroll
  for (int r = 0; r < 8; ++r) { rowmax[r] = -3.0e38f; rowsum[r] = 0.0f; }

  kload(0);
  kstore(0);
  __syncthreads();

  const int T = NSEQ / 32;   // 64 key tiles
  for (int t = 0; t < T; ++t) {
    const int cur = t & 1;

    if (t + 2 < T)   // prefetch two key tiles ahead
      __builtin_prefetch(qb + (size_t)((t + 2) * 32 + krow) * DMODEL + cb, 0, 1);
    if (t + 1 < T) kload((t + 1) * 32);

    // S = Q * K^T for 16 rows x 32 keys: 2 col-frags x 4 K-steps
    v8f s[2] = {};
    #pragma unroll
    for (int c = 0; c < 2; ++c) {
      const bf16_t* kp = &Kt[cur][c * 16 + ln][0];
      #pragma unroll
      for (int ks = 0; ks < 4; ++ks) {
        v16bf bfr = join8(kp + ks * 32 + hi * 8, kp + ks * 32 + 16 + hi * 8);
        s[c] = wmma_bf16(aq[ks], bfr, s[c]);
      }
    }

    // Online softmax (row stats reduced across 16-lane column groups)
    float p0[8], p1[8], corr[8];
    #pragma unroll
    for (int r = 0; r < 8; ++r) {
      float a0 = s[0][r] * scale;
      float a1 = s[1][r] * scale;
      float t2 = fmaxf(a0, a1);
      t2 = fmaxf(t2, __shfl_xor(t2, 1));
      t2 = fmaxf(t2, __shfl_xor(t2, 2));
      t2 = fmaxf(t2, __shfl_xor(t2, 4));
      t2 = fmaxf(t2, __shfl_xor(t2, 8));
      const float nm = fmaxf(rowmax[r], t2);
      corr[r]  = __expf(rowmax[r] - nm);
      rowmax[r] = nm;
      p0[r] = __expf(a0 - nm);
      p1[r] = __expf(a1 - nm);
      float ts = p0[r] + p1[r];
      ts += __shfl_xor(ts, 1);
      ts += __shfl_xor(ts, 2);
      ts += __shfl_xor(ts, 4);
      ts += __shfl_xor(ts, 8);
      rowsum[r] = rowsum[r] * corr[r] + ts;
    }
    #pragma unroll
    for (int f = 0; f < 8; ++f)
      #pragma unroll
      for (int r = 0; r < 8; ++r)
        acc[f][r] *= corr[r];

    // P (C layout) -> per-wave LDS tile (row-major 16x32)
    #pragma unroll
    for (int r = 0; r < 8; ++r) {
      Pl[wave][r + hi * 8][ln]      = (bf16_t)p0[r];
      Pl[wave][r + hi * 8][16 + ln] = (bf16_t)p1[r];
    }
    __builtin_amdgcn_wave_barrier();   // same-wave DS ops are in-order; block reordering

    // A-fragment of P (16x32, K = 32 keys)
    const bf16_t* pp = &Pl[wave][ln][0];
    v16bf ap = join8(pp + hi * 8, pp + 16 + hi * 8);

    // Y += P * Ktile   (B fragments from transposed tile: same key split as A)
    #pragma unroll
    for (int f = 0; f < 8; ++f) {
      const bf16_t* bp = &Ktt[cur][f * 16 + ln][0];
      v16bf bfr = join8(bp + hi * 8, bp + 16 + hi * 8);
      acc[f] = wmma_bf16(ap, bfr, acc[f]);
    }

    if (t + 1 < T) kstore(cur ^ 1);   // other buffer: safe vs readers of cur
    __syncthreads();
  }

  // Normalize and store Y (bf16, same [B,N,D] layout as Q)
  float inv[8];
  #pragma unroll
  for (int r = 0; r < 8; ++r) inv[r] = 1.0f / rowsum[r];
  #pragma unroll
  for (int f = 0; f < 8; ++f)
    #pragma unroll
    for (int r = 0; r < 8; ++r) {
      const int row = m0 + r + hi * 8;
      yb[(size_t)row * DMODEL + f * 16 + ln] = (bf16_t)(acc[f][r] * inv[r]);
    }
}

// ---------------------------------------------------------------------------
extern "C" void kernel_launch(void* const* d_in, const int* in_sizes, int n_in,
                              void* d_out, int out_size, void* d_ws, size_t ws_size,
                              hipStream_t stream) {
  (void)in_sizes; (void)n_in; (void)out_size; (void)ws_size;
  const float* x  = (const float*)d_in[0];
  const float* Wq = (const float*)d_in[1];
  const float* Wo = (const float*)d_in[2];
  const float* bo = (const float*)d_in[3];
  float* out = (float*)d_out;

  bf16_t* Qb = (bf16_t*)d_ws;                         // [B,N,D] bf16 (16 MB)
  bf16_t* Yb = Qb + (size_t)BDIM * NSEQ * DMODEL;     // [B,N,D] bf16 (16 MB)

  const int M = BDIM * NSEQ;                          // 4096
  dim3 gemmGrid(DMODEL / 128, M / 128);               // 16 x 32

  // Stage 1: Q = X * Wq  (fp32 in -> bf16 out)
  gemm_wmma<float, bf16_t, false><<<gemmGrid, 256, 0, stream>>>(
      x, Wq, nullptr, Qb, M, DMODEL, DMODEL);

  // Stage 2+3: flash attention per (b,h), 128 query rows per block
  attn_wmma<<<dim3(BDIM * NHEADS, NSEQ / 128), 256, 0, stream>>>(Qb, Yb);

  // Stage 4: Out = Y * Wo + bo  (bf16 in -> fp32 out)
  gemm_wmma<bf16_t, float, true><<<gemmGrid, 256, 0, stream>>>(
      Yb, Wo, bo, out, M, DMODEL, DMODEL);
}
